// Net_43267500540203
// MI455X (gfx1250) — matrix operand order repaired and checked by reading, compiled, not run on hardware
//
#include <hip/hip_runtime.h>
#include <hip/hip_bf16.h>

// ---------------------------------------------------------------------------
// GRCN forward for MI455X (gfx1250, wave32).
//  * GEMMs: v_wmma_f32_16x16x32_bf16, one wave per 16-row strip computing all
//    four 16-col tiles -> A (245MB for the 30000x2048 projection) is streamed
//    from HBM exactly once (~10.5us @ 23.3TB/s); W is pre-packed into
//    per-lane fragment order (32B contiguous per lane = 2x b128, coalesced).
//  * Graph ops: CSR built once on device (count + single-block scan + fill),
//    then GAT softmax / SAGE aggregation are gather kernels (wave per node,
//    lane-parallel over edges then feature-parallel accumulate) -> zero
//    floating-point atomics.
// ---------------------------------------------------------------------------

#define NUM_USER 50000
#define NUM_ITEM 30000
#define NNODES   80000
#define NEDGE    300000
#define DIM      64
#define SLOPE    0.01f
#define EPSN     1e-12f

typedef __attribute__((ext_vector_type(16))) __bf16 v16bf;
typedef __attribute__((ext_vector_type(8)))  float  v8f;

// ----------------------------- edge helpers --------------------------------

__device__ __forceinline__ void edge_sd(int e, const int* eu, const int* ei,
                                        int E, int doubled, int& s, int& d) {
  if (!doubled || e < E) { s = ei[e]; d = eu[e]; }         // items -> users
  else                   { s = eu[e - E]; d = ei[e - E]; } // users -> items
}

__device__ __forceinline__ int edge_src(int e, const int* eu, const int* ei,
                                        int E, int doubled) {
  return (!doubled || e < E) ? ei[e] : eu[e - E];
}

// ----------------------------- utility kernels -----------------------------

__global__ void ifill_k(int* p, int v, long n) {
  long i = (long)blockIdx.x * blockDim.x + threadIdx.x;
  if (i < n) p[i] = v;
}

__global__ void icopy_k(const int* in, int* out, long n) {
  long i = (long)blockIdx.x * blockDim.x + threadIdx.x;
  if (i < n) out[i] = in[i];
}

__global__ void copy_k(const float* in, float* out, long n) {
  long i = (long)blockIdx.x * blockDim.x + threadIdx.x;
  if (i < n) out[i] = in[i];
}

// row-wise l2norm, one wave per 64-wide row (2 elems/lane, shfl reduce)
__global__ void l2norm_k(const float* in, float* out, int R) {
  int wid  = (blockIdx.x * blockDim.x + threadIdx.x) >> 5;
  int lane = threadIdx.x & 31;
  if (wid >= R) return;
  const float* r = in + (size_t)wid * DIM;
  float x0 = r[lane], x1 = r[lane + 32];
  float p = x0 * x0 + x1 * x1;
  #pragma unroll
  for (int off = 16; off; off >>= 1) p += __shfl_xor(p, off);
  float inv = rsqrtf(p + EPSN);
  out[(size_t)wid * DIM + lane]      = x0 * inv;
  out[(size_t)wid * DIM + lane + 32] = x1 * inv;
}

// pref = l2norm(pref + h)
__global__ void addnorm_k(const float* pref, const float* h, float* out, int R) {
  int wid  = (blockIdx.x * blockDim.x + threadIdx.x) >> 5;
  int lane = threadIdx.x & 31;
  if (wid >= R) return;
  size_t b = (size_t)wid * DIM;
  float x0 = pref[b + lane]      + h[b + lane];
  float x1 = pref[b + lane + 32] + h[b + lane + 32];
  float p = x0 * x0 + x1 * x1;
  #pragma unroll
  for (int off = 16; off; off >>= 1) p += __shfl_xor(p, off);
  float inv = rsqrtf(p + EPSN);
  out[b + lane]      = x0 * inv;
  out[b + lane + 32] = x1 * inv;
}

// ------------------------------ CSR building -------------------------------

__global__ void count_k(const int* eu, const int* ei, int E, int doubled,
                        int* cnt, int NE) {
  int e = (int)(blockIdx.x * (long)blockDim.x + threadIdx.x);
  if (e >= NE) return;
  int s, d; edge_sd(e, eu, ei, E, doubled, s, d);
  atomicAdd(&cnt[d], 1);
}

// single-block chunked exclusive scan; out has n+1 entries
__global__ void scan_k(const int* in, int* out, int n) {
  __shared__ int buf[1024];
  __shared__ int carry;
  int tid = threadIdx.x;
  if (tid == 0) carry = 0;
  __syncthreads();
  for (int base = 0; base < n; base += 1024) {
    int i = base + tid;
    int v = (i < n) ? in[i] : 0;
    buf[tid] = v;
    __syncthreads();
    for (int ofs = 1; ofs < 1024; ofs <<= 1) {
      int t = (tid >= ofs) ? buf[tid - ofs] : 0;
      __syncthreads();
      buf[tid] += t;
      __syncthreads();
    }
    if (i < n) out[i] = carry + buf[tid] - v;
    __syncthreads();
    if (tid == 0) carry += buf[1023];
    __syncthreads();
  }
  if (tid == 0) out[n] = carry;
}

__global__ void fillcsr_k(const int* eu, const int* ei, int E, int doubled,
                          int* cur, int* eix, int NE) {
  int e = (int)(blockIdx.x * (long)blockDim.x + threadIdx.x);
  if (e >= NE) return;
  int s, d; edge_sd(e, eu, ei, E, doubled, s, d);
  int pos = atomicAdd(&cur[d], 1);
  eix[pos] = e;
}

// ------------------------------- WMMA GEMM ---------------------------------
// Pack W[K x 64] f32 -> bf16 fragments: for k-block kb32, n-tile nt, lane l,
// 16 contiguous bf16 = B-fragment of v_wmma_f32_16x16x32_bf16
// (lanes 0-15: K={0..7,16..23} col=lane; lanes 16-31: K={8..15,24..31} col=lane-16).
__global__ void pack_w_k(const float* W, __bf16* Wp, int K) {
  long idx = (long)blockIdx.x * blockDim.x + threadIdx.x;
  if (idx >= (long)K * DIM) return;
  int i = (int)(idx & 15); long t = idx >> 4;
  int lane = (int)(t & 31); t >>= 5;
  int nt = (int)(t & 3); int kblk = (int)(t >> 2);
  int kb = (lane < 16) ? 0 : 8;
  Wp[idx] = (__bf16)W[(size_t)(kblk * 32 + kb + (i < 8 ? i : i + 8)) * DIM + nt * 16 + (lane & 15)];
}

// C[M x 64] = act( A[M x K](f32) @ W(bf16 packed) + bias ). One wave per
// 16-row strip; 4 n-tile accumulators reuse the A fragment (A read once).
// A-fragment (ISA 7.12.2, 16-bit A 16x32): lanes 0-15 K={0..7,16..23},
// lanes 16-31 K={8..15,24..31}.
__global__ void gemm_wmma_k(const float* __restrict__ A,
                            const __bf16* __restrict__ Wp,
                            const float* __restrict__ bias,
                            float* __restrict__ C,
                            int M, int K, int act) {
  int wave = (int)((blockIdx.x * (long)blockDim.x + threadIdx.x) >> 5);
  int lane = threadIdx.x & 31;
  if (wave >= (M >> 4)) return;
  int row = (wave << 4) + (lane & 15);
  int ka = (lane < 16) ? 0 : 8;
  const float* arow = A + (size_t)row * K;
  v8f acc[4] = {v8f{}, v8f{}, v8f{}, v8f{}};
  for (int k = 0; k < K; k += 32) {
    __builtin_prefetch(arow + k + 256, 0, 0);   // global_prefetch_b8, 1KB ahead
    float4 a0 = *(const float4*)(arow + k + ka);
    float4 a1 = *(const float4*)(arow + k + ka + 4);
    float4 a2 = *(const float4*)(arow + k + 16 + ka);
    float4 a3 = *(const float4*)(arow + k + 16 + ka + 4);
    v16bf af;
    af[0]  = (__bf16)a0.x; af[1]  = (__bf16)a0.y; af[2]  = (__bf16)a0.z; af[3]  = (__bf16)a0.w;
    af[4]  = (__bf16)a1.x; af[5]  = (__bf16)a1.y; af[6]  = (__bf16)a1.z; af[7]  = (__bf16)a1.w;
    af[8]  = (__bf16)a2.x; af[9]  = (__bf16)a2.y; af[10] = (__bf16)a2.z; af[11] = (__bf16)a2.w;
    af[12] = (__bf16)a3.x; af[13] = (__bf16)a3.y; af[14] = (__bf16)a3.z; af[15] = (__bf16)a3.w;
    const v16bf* wp = (const v16bf*)(Wp + ((size_t)(k >> 5) * 2048)) + lane;
    #pragma unroll
    for (int nt = 0; nt < 4; ++nt) {
      v16bf bf = wp[nt * 32];                    // 32B contiguous per lane
      acc[nt] = __builtin_amdgcn_wmma_f32_16x16x32_bf16(
          false, af, false, bf, (short)0, acc[nt], false, false);
    }
  }
  // D layout: VGPR r -> M = strip + r + (lane<16 ? 0 : 8), N = nt*16 + lane%16
  int orow = (wave << 4) + ((lane < 16) ? 0 : 8);
  #pragma unroll
  for (int nt = 0; nt < 4; ++nt) {
    int col = nt * 16 + (lane & 15);
    float bb = bias[col];
    #pragma unroll
    for (int r = 0; r < 8; ++r) {
      float v = acc[nt][r] + bb;
      if (act) v = v > 0.f ? v : SLOPE * v;
      C[(size_t)(orow + r) * DIM + col] = v;
    }
  }
}

// ------------------------------- GAT kernels -------------------------------

// a[e] = <x[dst], x[src]>, one wave per edge (coalesced 256B row reads)
__global__ void edge_dot_k(const float* x, const int* eu, const int* ei,
                           int E, int doubled, float* a, int NE) {
  int wid  = (int)((blockIdx.x * (long)blockDim.x + threadIdx.x) >> 5);
  int lane = threadIdx.x & 31;
  if (wid >= NE) return;
  int s, d; edge_sd(wid, eu, ei, E, doubled, s, d);
  const float* xs = x + (size_t)s * DIM;
  const float* xd = x + (size_t)d * DIM;
  float p = xs[lane] * xd[lane] + xs[lane + 32] * xd[lane + 32];
  #pragma unroll
  for (int off = 16; off; off >>= 1) p += __shfl_down(p, off);
  if (lane == 0) a[wid] = p;
}

// GAT softmax + aggregation as a gather: one wave per destination node.
// Phase 1/2: lanes parallel over incident edges (max, sum-exp).
// Phase 3: sequential over edges (avg degree ~6-8), lanes parallel over DIM.
__global__ void gat_node_k(const float* __restrict__ x, const float* __restrict__ a,
                           const int* __restrict__ off, const int* __restrict__ eix,
                           const int* eu, const int* ei, int E, int doubled,
                           float* __restrict__ outH, float* alpha_out, int Nn) {
  int node = (int)((blockIdx.x * (long)blockDim.x + threadIdx.x) >> 5);
  int lane = threadIdx.x & 31;
  if (node >= Nn) return;
  int beg = off[node], end = off[node + 1];
  float mx = -INFINITY;
  for (int p = beg + lane; p < end; p += 32) mx = fmaxf(mx, a[eix[p]]);
  #pragma unroll
  for (int o = 16; o; o >>= 1) mx = fmaxf(mx, __shfl_xor(mx, o));
  float se = 0.f;
  for (int p = beg + lane; p < end; p += 32) se += expf(a[eix[p]] - mx);
  #pragma unroll
  for (int o = 16; o; o >>= 1) se += __shfl_xor(se, o);
  float inv = 1.f / (se + EPSN);
  float acc0 = 0.f, acc1 = 0.f;
  for (int p = beg; p < end; ++p) {
    int e = eix[p];
    float al = expf(a[e] - mx) * inv;
    if (alpha_out && lane == 0) alpha_out[e] = al;
    const float* xs = x + (size_t)edge_src(e, eu, ei, E, doubled) * DIM;
    acc0 += al * xs[lane];
    acc1 += al * xs[lane + 32];
  }
  outH[(size_t)node * DIM + lane]      = acc0;
  outH[(size_t)node * DIM + lane + 32] = acc1;
}

// SAGE weighted aggregation as gather: out[i] = sum_e w[e] * x[src(e)]
__global__ void sage_node_k(const float* __restrict__ x, const float* __restrict__ w,
                            const int* __restrict__ off, const int* __restrict__ eix,
                            const int* eu, const int* ei, int E,
                            float* __restrict__ outH, int Nn) {
  int node = (int)((blockIdx.x * (long)blockDim.x + threadIdx.x) >> 5);
  int lane = threadIdx.x & 31;
  if (node >= Nn) return;
  int beg = off[node], end = off[node + 1];
  float acc0 = 0.f, acc1 = 0.f;
  for (int p = beg; p < end; ++p) {
    int e = eix[p];
    float al = w[e];
    const float* xs = x + (size_t)edge_src(e, eu, ei, E, 1) * DIM;
    acc0 += al * xs[lane];
    acc1 += al * xs[lane + 32];
  }
  outH[(size_t)node * DIM + lane]      = acc0;
  outH[(size_t)node * DIM + lane + 32] = acc1;
}

// ------------------------------ fusion kernels -----------------------------

// rep = x + leaky(xh), written to d_out[:, coloff:coloff+64] (row stride 192)
__global__ void rep_k(const float* x, const float* xh, float* outp,
                      long n, int coloff) {
  long i = (long)blockIdx.x * blockDim.x + threadIdx.x;
  if (i >= n) return;
  long r = i >> 6; int c = (int)(i & 63);
  float v = xh[i]; v = v > 0.f ? v : SLOPE * v;
  outp[r * 192 + coloff + c] = x[i] + v;
}

// weight[e] = relu(max(alpha_v*conf[dst,0], alpha_a*conf[dst,1]))
__global__ void weight_k(const float* av, const float* aa, const float* conf,
                         const int* eu, const int* ei, int E, float* w, int NE) {
  int e = (int)(blockIdx.x * (long)blockDim.x + threadIdx.x);
  if (e >= NE) return;
  int ss, dd; edge_sd(e, eu, ei, E, 1, ss, dd);
  float wv = av[e] * conf[(size_t)dd * 2 + 0];
  float wa = aa[e] * conf[(size_t)dd * 2 + 1];
  float m = fmaxf(wv, wa);
  w[e] = m > 0.f ? m : 0.f;
}

// d_out[:, 0:64] = a + b + c
__global__ void add3_k(const float* a, const float* b, const float* c,
                       float* outp, long n) {
  long i = (long)blockIdx.x * blockDim.x + threadIdx.x;
  if (i >= n) return;
  long r = i >> 6; int cc = (int)(i & 63);
  outp[r * 192 + cc] = a[i] + b[i] + c[i];
}

// ------------------------------- host driver -------------------------------

static inline long cdiv(long a, long b) { return (a + b - 1) / b; }

extern "C" void kernel_launch(void* const* d_in, const int* in_sizes, int n_in,
                              void* d_out, int out_size, void* d_ws, size_t ws_size,
                              hipStream_t stream) {
  const int*   edge_u = (const int*)d_in[0];
  const int*   edge_i = (const int*)d_in[1];
  const float* v_feat = (const float*)d_in[2];
  const float* a_feat = (const float*)d_in[3];
  const float* pref_v = (const float*)d_in[4];
  const float* pref_a = (const float*)d_in[5];
  const float* Wv     = (const float*)d_in[6];
  const float* bv     = (const float*)d_in[7];
  const float* Wa     = (const float*)d_in[8];
  const float* ba     = (const float*)d_in[9];
  const float* id_emb = (const float*)d_in[10];
  const float* W1     = (const float*)d_in[11];
  const float* b1     = (const float*)d_in[12];
  const float* W2     = (const float*)d_in[13];
  const float* b2     = (const float*)d_in[14];
  const float* conf   = (const float*)d_in[15];
  float* out = (float*)d_out;

  const int U = NUM_USER, I = NUM_ITEM, N = NNODES, E = NEDGE;
  const int E2 = 2 * E;

  // bump allocator on d_ws
  char* ws = (char*)d_ws; size_t off = 0;
  auto alloc = [&](size_t bytes) -> void* {
    off = (off + 255) & ~(size_t)255;
    void* p = ws + off; off += bytes; return p;
  };
  float*  xB    = (float*)alloc((size_t)N * DIM * 4);   // concat(pref,f) / xn
  float*  hB    = (float*)alloc((size_t)N * DIM * 4);   // gat out / sage agg
  float*  prefB = (float*)alloc((size_t)U * DIM * 4);
  float*  aE    = (float*)alloc((size_t)E2 * 4);
  float*  alV   = (float*)alloc((size_t)E2 * 4);
  float*  alA   = (float*)alloc((size_t)E2 * 4);
  float*  wEb   = (float*)alloc((size_t)E2 * 4);
  __bf16* Wp    = (__bf16*)alloc((size_t)2048 * DIM * 2);
  float*  x1B   = (float*)alloc((size_t)N * DIM * 4);
  float*  x2B   = (float*)alloc((size_t)N * DIM * 4);
  int*    cntN  = (int*)alloc((size_t)N * 4);
  int*    curN  = (int*)alloc((size_t)N * 4);
  int*    offU  = (int*)alloc((size_t)(U + 1) * 4);     // routing CSR (dst=user)
  int*    eixU  = (int*)alloc((size_t)E * 4);
  int*    off2  = (int*)alloc((size_t)(N + 1) * 4);     // doubled CSR
  int*    eix2  = (int*)alloc((size_t)E2 * 4);

  const int BT = 256;
  auto copy = [&](const float* in, float* o, long n) {
    copy_k<<<cdiv(n, BT), BT, 0, stream>>>(in, o, n); };
  auto gemm = [&](const float* A, const float* Wf, const float* bias,
                  float* C, int M, int K, int act) {
    pack_w_k<<<cdiv((long)K * DIM, BT), BT, 0, stream>>>(Wf, Wp, K);
    gemm_wmma_k<<<cdiv((long)(M / 16) * 32, BT), BT, 0, stream>>>(A, Wp, bias, C, M, K, act);
  };
  auto l2n = [&](const float* in, float* o, int R) {
    l2norm_k<<<cdiv((long)R * 32, BT), BT, 0, stream>>>(in, o, R); };

  // ---- build both CSRs once ----
  auto build_csr = [&](int doubled, int NE, int Nn, int* offp, int* eixp) {
    ifill_k<<<cdiv(Nn, BT), BT, 0, stream>>>(cntN, 0, Nn);
    count_k<<<cdiv(NE, BT), BT, 0, stream>>>(edge_u, edge_i, E, doubled, cntN, NE);
    scan_k<<<1, 1024, 0, stream>>>(cntN, offp, Nn);
    icopy_k<<<cdiv(Nn, BT), BT, 0, stream>>>(offp, curN, Nn);
    fillcsr_k<<<cdiv(NE, BT), BT, 0, stream>>>(edge_u, edge_i, E, doubled, curN, eixp, NE);
  };
  build_csr(0, E,  U, offU, eixU);
  build_csr(1, E2, N, off2, eix2);

  // one GAT pass: attention dots then per-node softmax-aggregate (no atomics)
  auto run_gat = [&](const float* x, int doubled, int NE, const int* offp,
                     const int* eixp, int Nn, float* outH, float* alphaOut) {
    edge_dot_k<<<cdiv((long)NE * 32, BT), BT, 0, stream>>>(x, edge_u, edge_i, E, doubled, aE, NE);
    gat_node_k<<<cdiv((long)Nn * 32, BT), BT, 0, stream>>>(x, aE, offp, eixp, edge_u, edge_i,
                                                           E, doubled, outH, alphaOut, Nn);
  };

  // one cgcn modality: writes rep into d_out[:, coloff:], alpha into alphaOut
  auto run_cgcn = [&](const float* feat, int Kf, const float* Wm, const float* bm,
                      const float* pref0, float* alphaOut, int coloff) {
    gemm(feat, Wm, bm, xB + (size_t)U * DIM, I, Kf, /*leaky*/1); // f = leaky(feat@W+b)
    l2n(xB + (size_t)U * DIM, xB + (size_t)U * DIM, I);          // f = l2norm(f)
    l2n(pref0, prefB, U);                                        // pref = l2norm(pref)
    for (int r = 0; r < 3; ++r) {                                // routing
      copy(prefB, xB, (long)U * DIM);
      run_gat(xB, /*doubled*/0, E, offU, eixU, U, hB, nullptr);
      addnorm_k<<<cdiv((long)U * 32, BT), BT, 0, stream>>>(prefB, hB, prefB, U);
    }
    copy(prefB, xB, (long)U * DIM);                              // final x
    run_gat(xB, /*doubled*/1, E2, off2, eix2, N, hB, alphaOut);
    rep_k<<<cdiv((long)N * DIM, BT), BT, 0, stream>>>(xB, hB, out, (long)N * DIM, coloff);
  };

  // ---- two modalities ----
  run_cgcn(v_feat, 2048, Wv, bv, pref_v, alV, 64);
  run_cgcn(a_feat,  128, Wa, ba, pref_a, alA, 128);

  // ---- edge weights ----
  weight_k<<<cdiv(E2, BT), BT, 0, stream>>>(alV, alA, conf, edge_u, edge_i, E, wEb, E2);

  // ---- egcn ----
  l2n(id_emb, xB, N);                                            // xn
  sage_node_k<<<cdiv((long)N * 32, BT), BT, 0, stream>>>(xB, wEb, off2, eix2,
                                                         edge_u, edge_i, E, hB, N);
  gemm(hB, W1, b1, x1B, N, DIM, 1);                              // x1
  sage_node_k<<<cdiv((long)N * 32, BT), BT, 0, stream>>>(x1B, wEb, off2, eix2,
                                                         edge_u, edge_i, E, hB, N);
  gemm(hB, W2, b2, x2B, N, DIM, 1);                              // x2
  add3_k<<<cdiv((long)N * DIM, BT), BT, 0, stream>>>(xB, x1B, x2B, out, (long)N * DIM);
}
